// loss2_35038343201104
// MI455X (gfx1250) — matrix operand order repaired and checked by reading, not compiled, not run on hardware
//
#include <hip/hip_runtime.h>
#include <hip/hip_bf16.h>

typedef __attribute__((ext_vector_type(16))) __bf16        v16bf;
typedef __attribute__((ext_vector_type(8)))  float         v8f;
typedef __attribute__((ext_vector_type(4)))  unsigned int  v4u;

#define T_ROWS 4608
#define C_DIM  1024
#define GROUP  9
#define R_ROWS 2560          // 5 * (4608/9) distinct i-rows
#define MT_TILES 160         // R_ROWS / 16
#define NQ_TILES 72          // (4608/16) / 4  (each wave does 4 N-tiles = 64 cols)

__constant__ int c_right[5]  = {1, 3, 4, 5, 7};
__constant__ int c_posidx[9] = {0, 0, 0, 1, 2, 3, 0, 4, 0}; // only rightpos slots are read

__device__ __forceinline__ unsigned short f2bf_rne(float f) {
    unsigned int u = __float_as_uint(f);
    u = (u + 0x7FFFu + ((u >> 16) & 1u)) >> 16;
    return (unsigned short)u;
}

union BFrag {
    v4u   q[2];
    v16bf v;
};

// ---------------------------------------------------------------- kernel 1
// Row-wise: norm = max(||x_t||, 1e-8); xnbf[t,c] = bf16(x[t,c]/norm)
__global__ void k_normalize(const float* __restrict__ x,
                            unsigned short* __restrict__ xnbf) {
    __shared__ float red[256];
    const int t   = blockIdx.x;
    const int tid = threadIdx.x;
    const float* row = x + (size_t)t * C_DIM;
    float v0 = row[tid +   0];
    float v1 = row[tid + 256];
    float v2 = row[tid + 512];
    float v3 = row[tid + 768];
    red[tid] = v0*v0 + v1*v1 + v2*v2 + v3*v3;
    __syncthreads();
    for (int w = 128; w > 0; w >>= 1) {
        if (tid < w) red[tid] += red[tid + w];
        __syncthreads();
    }
    const float scale = 1.0f / fmaxf(sqrtf(red[0]), 1e-8f);
    unsigned short* orow = xnbf + (size_t)t * C_DIM;
    orow[tid +   0] = f2bf_rne(v0 * scale);
    orow[tid + 256] = f2bf_rne(v1 * scale);
    orow[tid + 512] = f2bf_rne(v2 * scale);
    orow[tid + 768] = f2bf_rne(v3 * scale);
}

// ---------------------------------------------------------------- kernel 2
// S[r, j] = sum_c xn[g(r), c] * xn[j, c]  (raw cosine, fp32)
// One wave -> 16 rows x 64 cols, K-loop of v_wmma_f32_16x16x32_bf16.
__global__ void __launch_bounds__(256)
k_gemm(const unsigned short* __restrict__ xnbf, float* __restrict__ S) {
    const int lane = threadIdx.x & 31;
    const int wave = threadIdx.x >> 5;
    const int wt   = blockIdx.x * 8 + wave;     // 0 .. MT_TILES*NQ_TILES-1
    const int mt   = wt % MT_TILES;
    const int nq   = wt / MT_TILES;

    // A fragment addressing (contiguous k order, same as B; not the ISA table order):
    //   lane L holds row (L&15); element e -> K = kb+e, kb=(L>>4)*16
    const int rloc = mt * 16 + (lane & 15);
    const int grow = (rloc / 5) * GROUP + c_right[rloc % 5];
    const unsigned short* arow = xnbf + (size_t)grow * C_DIM;
    const int kb = (lane >> 4) * 16;

    // B fragment addressing (same contiguous k order as A): lane L = column (L&15), K = (L>>4)*16 + e
    const int kbB = (lane >> 4) * 16;
    const int ncol = lane & 15;
    const unsigned short* b0 = xnbf + (size_t)(nq * 64 +  0 + ncol) * C_DIM + kbB;
    const unsigned short* b1 = xnbf + (size_t)(nq * 64 + 16 + ncol) * C_DIM + kbB;
    const unsigned short* b2 = xnbf + (size_t)(nq * 64 + 32 + ncol) * C_DIM + kbB;
    const unsigned short* b3 = xnbf + (size_t)(nq * 64 + 48 + ncol) * C_DIM + kbB;

    v8f acc0 = {}, acc1 = {}, acc2 = {}, acc3 = {};

    for (int k0 = 0; k0 < C_DIM; k0 += 32) {
        BFrag a, f0, f1, f2, f3;
        a.q[0]  = *(const v4u*)(arow + k0 + kb);
        a.q[1]  = *(const v4u*)(arow + k0 + kb + 8);
        f0.q[0] = *(const v4u*)(b0 + k0);
        f0.q[1] = *(const v4u*)(b0 + k0 + 8);
        f1.q[0] = *(const v4u*)(b1 + k0);
        f1.q[1] = *(const v4u*)(b1 + k0 + 8);
        f2.q[0] = *(const v4u*)(b2 + k0);
        f2.q[1] = *(const v4u*)(b2 + k0 + 8);
        f3.q[0] = *(const v4u*)(b3 + k0);
        f3.q[1] = *(const v4u*)(b3 + k0 + 8);

        acc0 = __builtin_amdgcn_wmma_f32_16x16x32_bf16(false, a.v, false, f0.v,
                                                       (short)0, acc0, false, false);
        acc1 = __builtin_amdgcn_wmma_f32_16x16x32_bf16(false, a.v, false, f1.v,
                                                       (short)0, acc1, false, false);
        acc2 = __builtin_amdgcn_wmma_f32_16x16x32_bf16(false, a.v, false, f2.v,
                                                       (short)0, acc2, false, false);
        acc3 = __builtin_amdgcn_wmma_f32_16x16x32_bf16(false, a.v, false, f3.v,
                                                       (short)0, acc3, false, false);
    }

    // C/D layout: VGPR v -> row (lane>>4)*8 + v, col = lane&15
    const int mrow0 = mt * 16 + (lane >> 4) * 8;
    const int cbase = nq * 64 + (lane & 15);
    #pragma unroll
    for (int v = 0; v < 8; ++v) {
        const size_t rb = (size_t)(mrow0 + v) * T_ROWS;
        S[rb + cbase +  0] = acc0[v];
        S[rb + cbase + 16] = acc1[v];
        S[rb + cbase + 32] = acc2[v];
        S[rb + cbase + 48] = acc3[v];
    }
}

// ---------------------------------------------------------------- kernel 3
// pl[p] = log( sum_k exp(S[r, neg_idx[p,k]]) ) - S[r, pair_j[p]]
__global__ void k_pairloss(const float* __restrict__ S,
                           const int* __restrict__ pair_i,
                           const int* __restrict__ pair_j,
                           const int* __restrict__ neg_idx,
                           float* __restrict__ pl, int K) {
    __shared__ float red[256];
    const int p   = blockIdx.x;
    const int tid = threadIdx.x;
    const int i   = pair_i[p];
    const int r   = (i / GROUP) * 5 + c_posidx[i % GROUP];
    const float* srow = S + (size_t)r * T_ROWS;
    const int* ni = neg_idx + (size_t)p * K;
    float s = 0.0f;
    for (int k = tid; k < K; k += 256) s += expf(srow[ni[k]]);
    red[tid] = s;
    __syncthreads();
    for (int w = 128; w > 0; w >>= 1) {
        if (tid < w) red[tid] += red[tid + w];
        __syncthreads();
    }
    if (tid == 0) pl[p] = logf(red[0]) - srow[pair_j[p]];
}

// ---------------------------------------------------------------- kernel 4
// out = (sum_p pl[p]) / (B * kuai2 * (kuai2-1) / 2), deterministic order
__global__ void k_reduce(const float* __restrict__ pl, int P, int nBlocks,
                         const int* __restrict__ kuai2p, float* __restrict__ out) {
    __shared__ float red[256];
    const int tid = threadIdx.x;
    float s = 0.0f;
    for (int k = tid; k < P; k += 256) s += pl[k];
    red[tid] = s;
    __syncthreads();
    for (int w = 128; w > 0; w >>= 1) {
        if (tid < w) red[tid] += red[tid + w];
        __syncthreads();
    }
    if (tid == 0) {
        const int k2 = kuai2p[0];
        const float denom = (float)nBlocks * (float)k2 * (float)(k2 - 1) * 0.5f;
        out[0] = red[0] / denom;
    }
}

// ---------------------------------------------------------------- launch
extern "C" void kernel_launch(void* const* d_in, const int* in_sizes, int n_in,
                              void* d_out, int out_size, void* d_ws, size_t ws_size,
                              hipStream_t stream) {
    const float* x        = (const float*)d_in[0];
    const int*   pair_i   = (const int*)d_in[1];
    const int*   pair_j   = (const int*)d_in[2];
    const int*   neg_idx  = (const int*)d_in[3];
    const int*   kuai2    = (const int*)d_in[4];

    const int P = in_sizes[1];                 // 10240
    const int K = in_sizes[3] / P;             // 1023
    const int nBlocks = T_ROWS / GROUP;        // 512

    char* ws = (char*)d_ws;
    unsigned short* xnbf = (unsigned short*)ws;                            //  9,437,184 B
    float*          S    = (float*)(ws + (size_t)T_ROWS * C_DIM * 2);      // 47,185,920 B
    float*          pl   = (float*)(ws + (size_t)T_ROWS * C_DIM * 2
                                       + (size_t)R_ROWS * T_ROWS * 4);     //     40,960 B

    k_normalize<<<T_ROWS, 256, 0, stream>>>(x, xnbf);
    k_gemm<<<(MT_TILES * NQ_TILES) / 8, 256, 0, stream>>>(xnbf, S);
    k_pairloss<<<P, 256, 0, stream>>>(S, pair_i, pair_j, neg_idx, pl, K);
    k_reduce<<<1, 256, 0, stream>>>(pl, P, nBlocks, kuai2, (float*)d_out);
}
